// Discriminator_46016279610164
// MI455X (gfx1250) — compile-verified, matrix operand order checked by reading, not run on hardware
//
#include <hip/hip_runtime.h>

typedef __bf16 bf16;
typedef __bf16 v16bf __attribute__((ext_vector_type(16)));
typedef float  v8f   __attribute__((ext_vector_type(8)));

__device__ __forceinline__ v8f wmma_bf16(v16bf a, v16bf b, v8f c) {
  // D = A*B + C, 16x16x32 bf16 -> f32, wave32
  return __builtin_amdgcn_wmma_f32_16x16x32_bf16(false, a, false, b, (short)0, c,
                                                 false, false);
}

__device__ __forceinline__ float sigmoidf_(float x) { return 1.f / (1.f + __expf(-x)); }

// CDNA5 async copy: global -> LDS directly (no VGPR round trip), ASYNCcnt-tracked.
__device__ __forceinline__ void async_g2l_b128(uint32_t lds_off, const void* gptr) {
  asm volatile("global_load_async_to_lds_b128 %0, %1, off" ::"v"(lds_off), "v"(gptr)
               : "memory");
}
__device__ __forceinline__ void wait_asynccnt0() {
  asm volatile("s_wait_asynccnt 0x0" ::: "memory");
}

// ---------------------------------------------------------------- utilities
__global__ void k_cvt(const float* __restrict__ s, bf16* __restrict__ d, size_t n) {
  size_t i = (size_t)blockIdx.x * blockDim.x + threadIdx.x;
  size_t st = (size_t)gridDim.x * blockDim.x;
  for (; i < n; i += st) d[i] = (bf16)s[i];
}

__global__ void k_pad_cvt(const float* __restrict__ s, bf16* __restrict__ d,
                          int rows, int sc, int dc) {
  size_t n = (size_t)rows * dc;
  size_t i = (size_t)blockIdx.x * blockDim.x + threadIdx.x;
  size_t st = (size_t)gridDim.x * blockDim.x;
  for (; i < n; i += st) {
    int r = (int)(i / dc);
    int c = (int)(i - (size_t)r * dc);
    d[i] = (c < sc) ? (bf16)s[(size_t)r * sc + c] : (bf16)0.f;
  }
}

__global__ void k_zero(uint32_t* __restrict__ p, size_t n) {
  size_t i = (size_t)blockIdx.x * blockDim.x + threadIdx.x;
  size_t st = (size_t)gridDim.x * blockDim.x;
  for (; i < n; i += st) p[i] = 0u;
}

__global__ void k_scale(float* __restrict__ p, int n, float sc) {
  int i = blockIdx.x * blockDim.x + threadIdx.x;
  if (i < n) p[i] *= sc;
}

// ------------------------------------------------- generic bf16 WMMA GEMM
// C[m, coff+n] = act( A(MxK) @ W(NxK)^T + bias[n] ), bf16 in/out, f32 acc.
// One 16x16 tile per wave, 8 waves per workgroup.
__global__ void __launch_bounds__(256)
k_gemm_bf16(const bf16* __restrict__ A, const bf16* __restrict__ W,
            const float* __restrict__ bias, bf16* __restrict__ C,
            int M, int N, int K, int ldc, int coff, int leaky) {
  const int tile = blockIdx.x * 8 + (threadIdx.x >> 5);
  const int ntn = N >> 4;
  const int mt = tile / ntn;
  const int nt = tile - mt * ntn;
  if (mt >= (M >> 4)) return;  // wave-uniform
  const int lane = threadIdx.x & 31;
  const int lr = lane & 15;
  const int lh = (lane >> 4) << 4;  // K-half select per lane group
  const bf16* ap = A + (size_t)(mt * 16 + lr) * K + lh;
  const bf16* wp = W + (size_t)(nt * 16 + lr) * K + lh;
  v8f acc = {};
  for (int k = 0; k < K; k += 32) {
    __builtin_prefetch(ap + k + 256, 0, 1);  // global_prefetch_b8
    __builtin_prefetch(wp + k + 256, 0, 1);
    acc = wmma_bf16(*(const v16bf*)(ap + k), *(const v16bf*)(wp + k), acc);
  }
  const int col = nt * 16 + lr;
  const float bv = bias ? bias[col] : 0.f;
  const int row0 = mt * 16 + ((lane >> 4) << 3);
#pragma unroll
  for (int v = 0; v < 8; ++v) {
    float x = acc[v] + bv;
    if (leaky) x = (x >= 0.f) ? x : 0.2f * x;
    C[(size_t)(row0 + v) * ldc + coff + col] = (bf16)x;
  }
}

// --------------------------------------------- persistent bidirectional GRU layer
// One launch per layer: grid (32,1,2) = 64 workgroups (trivially co-resident on
// MI455X), T-step loop inside the kernel, inter-workgroup sync per direction via
// an agent-scope atomic counter barrier. Per step each wave owns a 16x16 column
// tile of H and runs the fused gi+gh WMMA chains (K = H + din), then the gate
// math and h-update in registers. A-tiles staged global->LDS with the CDNA5
// async copy path (ASYNCcnt). h state ping-pongs between steps.
__global__ void __launch_bounds__(256)
k_gru_layer(const bf16* __restrict__ x,  // (B, T, din)
            const bf16* __restrict__ w_ih_f, const bf16* __restrict__ w_hh_f,
            const float* __restrict__ b_ih_f, const float* __restrict__ b_hh_f,
            const bf16* __restrict__ w_ih_b, const bf16* __restrict__ w_hh_b,
            const float* __restrict__ b_ih_b, const float* __restrict__ b_hh_b,
            float* __restrict__ hf,      // [ping][dir][B][H] f32 state
            bf16* __restrict__ hbuf,     // [ping][dir][B][H] bf16 state
            bf16* __restrict__ y,        // (B, T, 2H) concat out, or null
            float* __restrict__ out_acc, // (B,) mean accumulator, or null
            uint32_t* __restrict__ ctr,  // [dir] barrier counters (pre-zeroed)
            int T, int din) {
  constexpr int H = 1024, Bb = 64;
  const int dir = blockIdx.z;
  const bf16* w_ih = dir ? w_ih_b : w_ih_f;
  const bf16* w_hh = dir ? w_hh_b : w_hh_f;
  const float* b_ih = dir ? b_ih_b : b_ih_f;
  const float* b_hh = dir ? b_hh_b : b_hh_f;

  extern __shared__ bf16 smem[];
  bf16* sh = smem;           // 16 x H   (h rows)
  bf16* sx = smem + 16 * H;  // 16 x din (x_t rows)

  const int mt = blockIdx.x & 3;                              // batch tile (B=64)
  const int nt = (blockIdx.x >> 2) * 8 + (threadIdx.x >> 5);  // H col tile 0..63
  const int lane = threadIdx.x & 31;
  const int lr = lane & 15;
  const int lh = (lane >> 4) << 4;

  const int n0 = nt * 16;
  const bf16* sa_h = sh + (size_t)lr * H + lh;
  const bf16* sa_x = sx + (size_t)lr * din + lh;
  const bf16* whr = w_hh + (size_t)(0 * H + n0 + lr) * H + lh;
  const bf16* whz = w_hh + (size_t)(1 * H + n0 + lr) * H + lh;
  const bf16* whn = w_hh + (size_t)(2 * H + n0 + lr) * H + lh;
  const bf16* wir = w_ih + (size_t)(0 * H + n0 + lr) * din + lh;
  const bf16* wiz = w_ih + (size_t)(1 * H + n0 + lr) * din + lh;
  const bf16* win = w_ih + (size_t)(2 * H + n0 + lr) * din + lh;

  const int col = n0 + lr;
  const float br = b_ih[col] + b_hh[col];
  const float bz = b_ih[H + col] + b_hh[H + col];
  const float bin = b_ih[2 * H + col];
  const float bhn = b_hh[2 * H + col];
  const int row0 = mt * 16 + ((lane >> 4) << 3);
  const int nxr = din / 8;  // 16B chunks per x row

  for (int s = 0; s < T; ++s) {
    const int t = dir ? (T - 1 - s) : s;
    const size_t pofs = (size_t)(((s & 1) << 1) + dir) * Bb * H;
    const size_t nofs = (size_t)((((s + 1) & 1) << 1) + dir) * Bb * H;
    const float* hfp = hf + pofs;
    const bf16* hbp = hbuf + pofs;
    float* hfn = hf + nofs;
    bf16* hbn = hbuf + nofs;

    // ---- async global->LDS staging of the 16-row A tiles ----
    {
      const bf16* hsrc = hbp + (size_t)(mt * 16) * H;
      for (int i = threadIdx.x; i < 16 * H / 8; i += 256)
        async_g2l_b128((uint32_t)(uintptr_t)(sh + (size_t)i * 8),
                       hsrc + (size_t)i * 8);
      for (int i = threadIdx.x; i < 16 * nxr; i += 256) {
        const int r = i / nxr, c = i - r * nxr;
        async_g2l_b128((uint32_t)(uintptr_t)(sx + (size_t)r * din + (size_t)c * 8),
                       x + ((size_t)(mt * 16 + r) * T + t) * din + (size_t)c * 8);
      }
      wait_asynccnt0();
      __syncthreads();
    }

    // ---- fused gh + gi WMMA chains ----
    v8f ar = {}, az = {}, ahn = {}, agn = {};
#pragma unroll 4
    for (int k = 0; k < H; k += 32) {  // gh = h_{t-1} @ w_hh^T
      v16bf a = *(const v16bf*)(sa_h + k);
      ar = wmma_bf16(a, *(const v16bf*)(whr + k), ar);
      az = wmma_bf16(a, *(const v16bf*)(whz + k), az);
      ahn = wmma_bf16(a, *(const v16bf*)(whn + k), ahn);
    }
#pragma unroll 4
    for (int k = 0; k < din; k += 32) {  // gi = x_t @ w_ih^T (fused)
      v16bf a = *(const v16bf*)(sa_x + k);
      ar = wmma_bf16(a, *(const v16bf*)(wir + k), ar);
      az = wmma_bf16(a, *(const v16bf*)(wiz + k), az);
      agn = wmma_bf16(a, *(const v16bf*)(win + k), agn);
    }

    // ---- gate math + h update (fp32) ----
#pragma unroll
    for (int v = 0; v < 8; ++v) {
      const int b = row0 + v;
      const float r = sigmoidf_(ar[v] + br);
      const float z = sigmoidf_(az[v] + bz);
      const float n = tanhf(agn[v] + bin + r * (ahn[v] + bhn));
      const float hp = hfp[(size_t)b * H + col];
      const float hnew = (1.f - z) * n + z * hp;
      hfn[(size_t)b * H + col] = hnew;
      hbn[(size_t)b * H + col] = (bf16)hnew;
      if (y) y[((size_t)b * T + t) * (2 * H) + (size_t)dir * H + col] = (bf16)hnew;
      if (out_acc) atomicAdd(&out_acc[b], hnew);
    }

    // ---- inter-workgroup barrier (per direction, 32 blocks) ----
    __threadfence();   // make this thread's h writes device-visible
    __syncthreads();   // whole block done before signaling
    if (threadIdx.x == 0)
      __hip_atomic_fetch_add(&ctr[dir], 1u, __ATOMIC_RELEASE,
                             __HIP_MEMORY_SCOPE_AGENT);
    const uint32_t target = 32u * (uint32_t)(s + 1);
    while (__hip_atomic_load(&ctr[dir], __ATOMIC_ACQUIRE,
                             __HIP_MEMORY_SCOPE_AGENT) < target)
      __builtin_amdgcn_s_sleep(2);
  }
}

// ---------------------------------------------------------------- launcher
extern "C" void kernel_launch(void* const* d_in, const int* in_sizes, int n_in,
                              void* d_out, int out_size, void* d_ws, size_t ws_size,
                              hipStream_t stream) {
  (void)in_sizes; (void)n_in; (void)out_size; (void)ws_size;
  constexpr int Bb = 64, T = 300, OUT = 72, H = 1024;
  constexpr int BT = Bb * T;

  const float* cond = (const float*)d_in[0];
  const float* input = (const float*)d_in[1];
  const float* ce_w1 = (const float*)d_in[2];
  const float* ce_b1 = (const float*)d_in[3];
  const float* ce_w2 = (const float*)d_in[4];
  const float* ce_b2 = (const float*)d_in[5];
  const float* me_w1 = (const float*)d_in[6];
  const float* me_b1 = (const float*)d_in[7];
  const float* me_w2 = (const float*)d_in[8];
  const float* me_b2 = (const float*)d_in[9];
  // order: l0f, l0b, l1f, l1b
  const float* w_ih[4] = {(const float*)d_in[10], (const float*)d_in[14],
                          (const float*)d_in[18], (const float*)d_in[22]};
  const float* w_hh[4] = {(const float*)d_in[11], (const float*)d_in[15],
                          (const float*)d_in[19], (const float*)d_in[23]};
  const float* b_ih[4] = {(const float*)d_in[12], (const float*)d_in[16],
                          (const float*)d_in[20], (const float*)d_in[24]};
  const float* b_hh[4] = {(const float*)d_in[13], (const float*)d_in[17],
                          (const float*)d_in[21], (const float*)d_in[25]};
  float* out = (float*)d_out;

  char* ws = (char*)d_ws;
  size_t off = 0;
  auto alloc = [&](size_t bytes) -> void* {
    void* p = ws + off;
    off += (bytes + 255) & ~(size_t)255;
    return p;
  };

  bf16* x0 = (bf16*)alloc((size_t)BT * 1024 * 2);  // encoder out / layer0 in
  bf16* x1 = (bf16*)alloc((size_t)BT * 2048 * 2);  // layer0 out / layer1 in
  bf16* cb = (bf16*)alloc((size_t)BT * 64 * 2);
  bf16* ib = (bf16*)alloc((size_t)BT * 96 * 2);    // input padded 72 -> 96
  bf16* h1c = (bf16*)alloc((size_t)BT * 256 * 2);
  bf16* h1m = (bf16*)alloc((size_t)BT * 256 * 2);
  bf16* cw1 = (bf16*)alloc(256 * 64 * 2);
  bf16* cw2 = (bf16*)alloc(512 * 256 * 2);
  bf16* mw1 = (bf16*)alloc(256 * 96 * 2);
  bf16* mw2 = (bf16*)alloc(512 * 256 * 2);
  bf16* wihb[4];
  bf16* whhb[4];
  const int din_l[4] = {1024, 1024, 2048, 2048};
  for (int i = 0; i < 4; ++i) {
    wihb[i] = (bf16*)alloc((size_t)3072 * din_l[i] * 2);
    whhb[i] = (bf16*)alloc((size_t)3072 * 1024 * 2);
  }
  float* hf = (float*)alloc((size_t)2 * 2 * Bb * H * 4);  // [ping][dir][B][H]
  bf16* hb = (bf16*)alloc((size_t)2 * 2 * Bb * H * 2);
  uint32_t* ctr = (uint32_t*)alloc(4 * sizeof(uint32_t));  // [layer][dir]

  auto cvt = [&](const float* s, bf16* d, size_t n) {
    int blocks = (int)((n + 255) / 256);
    if (blocks > 4096) blocks = 4096;
    k_cvt<<<blocks, 256, 0, stream>>>(s, d, n);
  };

  // ---- fp32 -> bf16 conversions (weights become L2-resident) ----
  cvt(cond, cb, (size_t)BT * 64);
  k_pad_cvt<<<4096, 256, 0, stream>>>(input, ib, BT, OUT, 96);
  cvt(ce_w1, cw1, 256 * 64);
  cvt(ce_w2, cw2, 512 * 256);
  k_pad_cvt<<<64, 256, 0, stream>>>(me_w1, mw1, 256, OUT, 96);
  cvt(me_w2, mw2, 512 * 256);
  for (int i = 0; i < 4; ++i) {
    cvt(w_ih[i], wihb[i], (size_t)3072 * din_l[i]);
    cvt(w_hh[i], whhb[i], (size_t)3072 * 1024);
  }

  auto gemm = [&](const bf16* A, const bf16* W, const float* bias, bf16* C, int M,
                  int N, int K, int ldc, int coff, int leaky) {
    int tiles = (M / 16) * (N / 16);
    k_gemm_bf16<<<tiles / 8, 256, 0, stream>>>(A, W, bias, C, M, N, K, ldc, coff,
                                               leaky);
  };
  // ---- encoders: x0 = [cond_mlp | input_mlp] ----
  gemm(cb, cw1, ce_b1, h1c, BT, 256, 64, 256, 0, 1);
  gemm(h1c, cw2, ce_b2, x0, BT, 512, 256, 1024, 0, 0);
  gemm(ib, mw1, me_b1, h1m, BT, 256, 96, 256, 0, 1);
  gemm(h1m, mw2, me_b2, x0, BT, 512, 256, 1024, 512, 0);

  k_zero<<<1, 64, 0, stream>>>((uint32_t*)out, 64);
  k_zero<<<1, 64, 0, stream>>>(ctr, 4);

  for (int layer = 0; layer < 2; ++layer) {
    k_zero<<<512, 256, 0, stream>>>((uint32_t*)hf, (size_t)2 * 2 * Bb * H);
    k_zero<<<256, 256, 0, stream>>>((uint32_t*)hb, (size_t)2 * 2 * Bb * H / 2);
    const int din = layer ? 2048 : 1024;
    const bf16* xin = layer ? x1 : x0;
    bf16* yout = layer ? nullptr : x1;
    float* oacc = layer ? out : nullptr;
    const int f = layer ? 2 : 0, bk = layer ? 3 : 1;
    const size_t shmem = (size_t)16 * (H + din) * 2;
    k_gru_layer<<<dim3(32, 1, 2), 256, shmem, stream>>>(
        xin, wihb[f], whhb[f], b_ih[f], b_hh[f], wihb[bk], whhb[bk], b_ih[bk],
        b_hh[bk], hf, hb, yout, oacc, ctr + 2 * layer, T, din);
  }
  k_scale<<<1, 64, 0, stream>>>(out, 64, 1.f / (300.f * 2048.f));
}
